// SparseStudentSNN1d_67834713473593
// MI455X (gfx1250) — 16-bit and 8-bit matrix operand order repaired and checked by reading, compiled, not run on hardware
//
#include <hip/hip_runtime.h>
#include <hip/hip_bf16.h>
#include <math.h>

typedef _Float16 f16;
typedef _Float16 v8h  __attribute__((ext_vector_type(8)));
typedef _Float16 v16h __attribute__((ext_vector_type(16)));
typedef float    v8f  __attribute__((ext_vector_type(8)));
typedef int      v4i  __attribute__((ext_vector_type(4)));
typedef int      v8i  __attribute__((ext_vector_type(8)));
typedef int      v16i __attribute__((ext_vector_type(16)));

#define T_STEPS 20
#define L_TOT   4096
#define CIN     12
#define LTILE   64
#define NXROWS  84
#define XPITCH  24
#define NSROWS  80
#define SPITCH8 80

__device__ inline v16h ld32B(const f16* p) {
    union { v16h v; v8h h[2]; } u;
    u.h[0] = *(const v8h*)p;
    u.h[1] = *(const v8h*)(p + 8);
    return u.v;
}

#if __has_builtin(__builtin_amdgcn_cvt_pk_fp8_f32)
#define HAVE_HW_FP8 1
#endif

__device__ inline unsigned int fp8_e4m3_sw(float f) {
    unsigned int u = __float_as_uint(f);
    unsigned int s = (u >> 24) & 0x80u;
    float a = fabsf(f);
    if (!(a < 448.0f)) return s | 0x7Eu;
    if (a == 0.0f) return s;
    int e; (void)frexpf(a, &e);
    int lsb = (e - 1) - 3;
    if (lsb < -9) lsb = -9;
    float sc = exp2f((float)(-lsb));
    float q = rintf(a * sc) / sc;
    if (q == 0.0f) return s;
    (void)frexpf(q, &e);
    int E = e - 1 + 7;
    unsigned int bits;
    if (E >= 1) {
        float m = q * exp2f((float)(-(e - 1))) - 1.0f;
        bits = ((unsigned)E << 3) | (unsigned)(m * 8.0f + 0.5f);
    } else {
        bits = (unsigned)(q * 512.0f + 0.5f);
    }
    if (bits > 0x7Eu) bits = 0x7Eu;
    return s | bits;
}

__device__ inline int pack4_e4m3(float f0, float f1, float f2, float f3) {
#ifdef HAVE_HW_FP8
    int w = __builtin_amdgcn_cvt_pk_fp8_f32(f0, f1, 0, false);
    w = __builtin_amdgcn_cvt_pk_fp8_f32(f2, f3, w, true);
    return w;
#else
    return (int)(fp8_e4m3_sw(f0) | (fp8_e4m3_sw(f1) << 8) |
                 (fp8_e4m3_sw(f2) << 16) | (fp8_e4m3_sw(f3) << 24));
#endif
}

__global__ __launch_bounds__(256) void snn_main(
    const float* __restrict__ x,
    const float* __restrict__ w1, const float* __restrict__ b1,
    const float* __restrict__ w2, const float* __restrict__ b2,
    const float* __restrict__ gainp, const float* __restrict__ th1p,
    const float* __restrict__ th2p,
    const float* __restrict__ fcw_g,
    float* __restrict__ ws_part)
{
    __shared__ __attribute__((aligned(16))) f16 xT[T_STEPS][NXROWS][XPITCH];
    __shared__ __attribute__((aligned(16))) unsigned char s1_8[NSROWS][SPITCH8];
    __shared__ float bias1[64], bias2[64];
    __shared__ float fcw[4 * 64];
    __shared__ float wavePart[8][2][4];

    const int tid   = threadIdx.x;
    const int lane  = tid & 31;
    const int wave  = tid >> 5;
    const int n16   = lane & 15;
    const int khalf = lane >> 4;
    const int bIdx  = blockIdx.y;
    const int l0    = blockIdx.x * LTILE;

    const float gain = gainp[0];
    const float th1  = th1p[0];
    const float th2  = th2p[0];
    const float inv_tau2 = 1.0f / 0.9f;

    const int mtW = wave >> 1;

    v16h a1[2];
    {
        const int m1 = mtW * 16 + n16;
        #pragma unroll
        for (int kc = 0; kc < 2; ++kc) {
            #pragma unroll
            for (int j = 0; j < 16; ++j) {
                int klocal = khalf * 16 + j;
                int k  = kc * 32 + klocal;
                int kk = k >> 4, c = k & 15;
                float v = (kk < 3 && c < CIN) ? w1[(m1 * CIN + c) * 3 + kk] : 0.0f;
                a1[kc][j] = (f16)v;
            }
        }
    }
    v16i a2k0;
    v8i  a2k1;
    {
        const int m2 = mtW * 16 + n16;
        #pragma unroll
        for (int v = 0; v < 16; ++v) {
            float fv[4];
            #pragma unroll
            for (int bp = 0; bp < 4; ++bp) {
                int vv = v & 7;
                int klocal = (v >> 3) * 64 + (vv >> 2) * 32 + khalf * 16 +
                             (vv & 3) * 4 + bp;
                int kk = klocal >> 6, c = klocal & 63;
                fv[bp] = w2[(m2 * 64 + c) * 3 + kk];
            }
            a2k0[v] = pack4_e4m3(fv[0], fv[1], fv[2], fv[3]);
        }
        #pragma unroll
        for (int v = 0; v < 8; ++v) {
            float fv[4];
            #pragma unroll
            for (int bp = 0; bp < 4; ++bp) {
                int c = (v >> 2) * 32 + khalf * 16 + (v & 3) * 4 + bp;
                fv[bp] = w2[(m2 * 64 + c) * 3 + 2];
            }
            a2k1[v] = pack4_e4m3(fv[0], fv[1], fv[2], fv[3]);
        }
    }

    for (int i = tid; i < 64; i += 256) { bias1[i] = b1[i]; bias2[i] = b2[i]; }
    for (int i = tid; i < 256; i += 256) fcw[i] = fcw_g[i];

    for (int p = tid; p < 16 * NXROWS; p += 256) {
        int c  = p / NXROWS;
        int hx = p - c * NXROWS;
        long gl = (long)l0 - 9 + hx;
        if (c < CIN && gl >= 0 && gl < L_TOT) {
            const float* src = x + (((long)bIdx * CIN + c) * L_TOT + gl) * T_STEPS;
            #pragma unroll
            for (int t = 0; t < T_STEPS; ++t)
                xT[t][hx][c] = (f16)src[t];
        } else {
            #pragma unroll
            for (int t = 0; t < T_STEPS; ++t)
                xT[t][hx][c] = (f16)0.0f;
        }
    }
    __syncthreads();

    v8f v2s[2] = {};
    v8f acc[2] = {};

    for (int tt = 0; tt < T_STEPS; ++tt) {
        {
            int ntLo = (wave & 1) ? 3 : 0;
            int ntHi = (wave & 1) ? 5 : 3;
            for (int nt = ntLo; nt < ntHi; ++nt) {
                v8f d = {0.f,0.f,0.f,0.f,0.f,0.f,0.f,0.f};
                #pragma unroll
                for (int kc = 0; kc < 2; ++kc) {
                    v16h bfr;
                    int kk = kc * 2 + khalf;
                    if (kk < 3) {
                        bfr = ld32B(&xT[tt][nt * 16 + n16 + kk][0]);
                    } else {
                        v16h z = {(f16)0,(f16)0,(f16)0,(f16)0,(f16)0,(f16)0,(f16)0,(f16)0,
                                  (f16)0,(f16)0,(f16)0,(f16)0,(f16)0,(f16)0,(f16)0,(f16)0};
                        bfr = z;
                    }
                    d = __builtin_amdgcn_wmma_f32_16x16x32_f16(
                            false, a1[kc], false, bfr, (short)0, d, false, false);
                }
                int hs = nt * 16 + n16;
                long gl = (long)l0 - 8 + hs;
                bool colOK = (gl >= 0 && gl < L_TOT);
                unsigned long long pk = 0ull;
                #pragma unroll
                for (int r = 0; r < 8; ++r) {
                    int c = mtW * 16 + r + khalf * 8;
                    float i1 = (d[r] + bias1[c]) * gain;
                    if (colOK && i1 >= th1) pk |= (0x38ull << (8 * r));
                }
                *(unsigned long long*)&s1_8[hs][mtW * 16 + khalf * 8] = pk;
            }
        }
        __syncthreads();

        #pragma unroll
        for (int q = 0; q < 2; ++q) {
            int nt = (wave * 2 + q) & 3;
            int rowb = nt * 16 + n16 + 7;
            union { v16i v; v4i s[4]; } b0;
            #pragma unroll
            for (int s = 0; s < 4; ++s) {
                int K  = s * 32 + khalf * 16;
                int kk = K >> 6, c0 = K & 63;
                b0.s[s] = *(const v4i*)&s1_8[rowb + kk][c0];
            }
            union { v8i v; v4i s[2]; } b1f;
            #pragma unroll
            for (int s = 0; s < 2; ++s) {
                int c0 = s * 32 + khalf * 16;
                b1f.s[s] = *(const v4i*)&s1_8[rowb + 2][c0];
            }
            v8f d = {0.f,0.f,0.f,0.f,0.f,0.f,0.f,0.f};
            d = __builtin_amdgcn_wmma_f32_16x16x128_fp8_fp8(
                    a2k0, b0.v, (short)0, d, false, false);
            d = __builtin_amdgcn_wmma_f32_16x16x64_fp8_fp8(
                    a2k1, b1f.v, (short)0, d, false, false);
            #pragma unroll
            for (int r = 0; r < 8; ++r) {
                int c = mtW * 16 + r + khalf * 8;
                float i2 = (d[r] + bias2[c]) * gain;
                float v  = v2s[q][r] + (i2 - v2s[q][r]) * inv_tau2;
                bool  sp = (v >= th2);
                acc[q][r] += sp ? 1.0f : 0.0f;
                v2s[q][r]  = sp ? 0.0f : v;
            }
        }
        __syncthreads();
    }

    #pragma unroll
    for (int q = 0; q < 2; ++q) {
        #pragma unroll
        for (int j = 0; j < 4; ++j) {
            float p = 0.0f;
            #pragma unroll
            for (int r = 0; r < 8; ++r) {
                int c = mtW * 16 + r + khalf * 8;
                p += acc[q][r] * fcw[j * 64 + c];
            }
            for (int off = 16; off >= 1; off >>= 1)
                p += __shfl_xor(p, off, 32);
            if (lane == 0) wavePart[wave][q][j] = p;
        }
    }
    __syncthreads();
    if (tid < 4) {
        float s = 0.0f;
        #pragma unroll
        for (int w = 0; w < 8; ++w)
            #pragma unroll
            for (int q = 0; q < 2; ++q)
                s += wavePart[w][q][tid];
        ws_part[((long)bIdx * 64 + blockIdx.x) * 4 + tid] = s;
    }
}

__global__ __launch_bounds__(32) void snn_reduce(
    const float* __restrict__ part, const float* __restrict__ fcb,
    float* __restrict__ out)
{
    int b = blockIdx.x;
    int j = threadIdx.x;
    if (j < 4) {
        float s = 0.0f;
        for (int tile = 0; tile < 64; ++tile)
            s += part[((long)b * 64 + tile) * 4 + j];
        out[b * 4 + j] = fcb[j] + s * (1.0f / (20.0f * 4096.0f));
    }
}

extern "C" void kernel_launch(void* const* d_in, const int* in_sizes, int n_in,
                              void* d_out, int out_size, void* d_ws, size_t ws_size,
                              hipStream_t stream) {
    const float* x    = (const float*)d_in[0];
    const float* w1   = (const float*)d_in[1];
    const float* b1   = (const float*)d_in[2];
    const float* w2   = (const float*)d_in[3];
    const float* b2   = (const float*)d_in[4];
    const float* gain = (const float*)d_in[5];
    const float* th1  = (const float*)d_in[6];
    const float* th2  = (const float*)d_in[7];
    const float* fcw  = (const float*)d_in[8];
    const float* fcb  = (const float*)d_in[9];
    float* out  = (float*)d_out;
    float* part = (float*)d_ws;

    dim3 grid(L_TOT / LTILE, 64);
    snn_main<<<grid, 256, 0, stream>>>(x, w1, b1, w2, b2, gain, th1, th2, fcw, part);
    snn_reduce<<<dim3(64), dim3(32), 0, stream>>>(part, fcb, out);
}
